// DecoderAttnBase_5368709120228
// MI455X (gfx1250) — matrix operand order repaired and checked by reading, not compiled, not run on hardware
//
#include <hip/hip_runtime.h>
#include <hip/hip_bf16.h>

// ---------------------------------------------------------------------------
// CDNA5 bf16 WMMA helpers (wave32).  D = A(16x32) * B(32x16) + C, f32 accum.
// ---------------------------------------------------------------------------
typedef __attribute__((ext_vector_type(16))) __bf16 v16bf;
typedef __attribute__((ext_vector_type(8)))  float  v8f;
typedef __attribute__((ext_vector_type(4)))  int    v4i;

__device__ __forceinline__ v8f wmma_bf16(v16bf a, v16bf b, v8f c) {
  return __builtin_amdgcn_wmma_f32_16x16x32_bf16(false, a, false, b,
                                                 (short)0, c, false, false);
}

// A operand: lane holds row m = lane&15, K = kb + (lane>>4)*16 + {0..15}
__device__ __forceinline__ v16bf a_from_f32(const float* __restrict__ p) {
  v16bf a;
#pragma unroll
  for (int j = 0; j < 16; ++j) a[j] = (__bf16)p[j];
  return a;
}
__device__ __forceinline__ v16bf a_from_lds(const __bf16* p) {
  v16bf a;
#pragma unroll
  for (int j = 0; j < 16; ++j) a[j] = p[j];
  return a;
}
// B operand: lane holds col n = lane&15, K = kb + (lane>>4)*16 + {0..15}
// == 16 contiguous elements of row n of a row-major weight W(N x K).
__device__ __forceinline__ v16bf b_from_f32(const float* __restrict__ p) {
  v16bf b;
#pragma unroll
  for (int j = 0; j < 16; ++j) b[j] = (__bf16)p[j];
  return b;
}
__device__ __forceinline__ v16bf b_from_lds(const __bf16* p) {
  v16bf b;
#pragma unroll
  for (int j = 0; j < 16; ++j) b[j] = p[j];
  return b;
}

__device__ __forceinline__ float sigmoidf(float x) { return 1.0f / (1.0f + __expf(-x)); }

// Optional gfx1250 async global->LDS staging path (guarded; safe if absent).
#if defined(__has_builtin)
#if __has_builtin(__builtin_amdgcn_global_load_async_to_lds_b128) && \
    __has_builtin(__builtin_amdgcn_s_wait_asynccnt)
#define USE_ASYNC_LDS 1
#endif
#endif
#ifdef USE_ASYNC_LDS
// Builtin wants v4i pointers in AS(1) (printed "__device__") / AS(3).
// Flat shared address: low 32 bits are the LDS offset (ISA 10.2 aperture map),
// so an integer round-trip produces a valid AS(3)/AS(1) pointer value.
#define AS3P(p) ((__attribute__((address_space(3))) v4i*)(unsigned int)(uintptr_t)(p))
#define AS1P(p) ((__attribute__((address_space(1))) v4i*)(uintptr_t)(p))
#endif

// Problem constants
#define CB   64      // batch
#define CS   128     // seq len
#define CH   1024    // hidden
#define CE   1024    // embed
#define CV   32000   // vocab

// ---------------------------------------------------------------------------
// K1: q_proj(64x1024) = h0 @ Wa_w^T + Wa_b.   grid (4,64) x 32 threads.
// ---------------------------------------------------------------------------
__global__ void qproj_kernel(const float* __restrict__ h0,
                             const float* __restrict__ Wa_w,
                             const float* __restrict__ Wa_b,
                             float* __restrict__ qp) {
  const int lane = threadIdx.x & 31, m = lane & 15, half = lane >> 4;
  const int row0 = blockIdx.x * 16, n0 = blockIdx.y * 16;
  v8f acc = {};
  for (int kb = 0; kb < CH; kb += 32) {
    v16bf a = a_from_f32(h0 + (row0 + m) * CH + kb + half * 16);
    v16bf b = b_from_f32(Wa_w + (size_t)(n0 + m) * CH + kb + half * 16);
    acc = wmma_bf16(a, b, acc);
  }
  const int col = n0 + m;
  const float bias = Wa_b[col];
#pragma unroll
  for (int v = 0; v < 8; ++v)
    qp[(row0 + v + 8 * half) * CH + col] = acc[v] + bias;
}

// ---------------------------------------------------------------------------
// K2: scores[b,s] = Va . tanh(k_proj[b,s,:] + Ua_b + q_proj[b,:]) + Va_b
// One block per (b, s-half of 64 rows): 128 blocks x 512 threads (16 waves).
// 64x1024 enc slab staged once as bf16 in LDS (~129 KB); each wave covers
// 4 N-tiles and keeps 4 M-tile accumulators so each B register-load feeds
// 4 WMMAs (4x less Ua_w traffic than 16-row blocks).
// ---------------------------------------------------------------------------
__global__ void scores_kernel(const float* __restrict__ enc,
                              const float* __restrict__ Ua_w,
                              const float* __restrict__ Ua_b,
                              const float* __restrict__ Va_w,
                              const float* __restrict__ Va_b,
                              const float* __restrict__ qp,
                              float* __restrict__ scores) {
  __shared__ __bf16 sA[64][1032];   // 64 s-rows x 1024 K (padded)  ~129 KB
  __shared__ float  sQ[CH];         // q_proj[b] + Ua_b
  __shared__ float  sV[CH];         // Va_w
  __shared__ float  sS[16][64];     // per-wave row partials

  const int tid = threadIdx.x;
  const int b = blockIdx.x >> 1, sh = blockIdx.x & 1;

  // stage enc[b, sh*64 .. +64, :] -> bf16 LDS (rows contiguous in memory)
  const float4* src = reinterpret_cast<const float4*>(enc + ((size_t)b * CS + sh * 64) * CH);
  for (int i = tid; i < 16384; i += 512) {
    float4 v = src[i];
    __bf16* d = &sA[i >> 8][(i & 255) << 2];
    d[0] = (__bf16)v.x; d[1] = (__bf16)v.y; d[2] = (__bf16)v.z; d[3] = (__bf16)v.w;
  }
  for (int c = tid; c < CH; c += 512) {
    sQ[c] = qp[b * CH + c] + Ua_b[c];
    sV[c] = Va_w[c];
  }
  __syncthreads();

  const int wave = tid >> 5, lane = tid & 31, m = lane & 15, half = lane >> 4;
  float ps[4][8];
#pragma unroll
  for (int st = 0; st < 4; ++st)
#pragma unroll
    for (int v = 0; v < 8; ++v) ps[st][v] = 0.0f;

  for (int nt = 0; nt < 4; ++nt) {
    const int n0 = (wave * 4 + nt) * 16;
    v8f acc[4];
#pragma unroll
    for (int st = 0; st < 4; ++st) acc[st] = (v8f){};
    for (int kb = 0; kb < CH; kb += 32) {
      v16bf bm = b_from_f32(Ua_w + (size_t)(n0 + m) * CH + kb + half * 16);
#pragma unroll
      for (int st = 0; st < 4; ++st) {
        v16bf a = a_from_lds(&sA[st * 16 + m][kb + half * 16]);
        acc[st] = wmma_bf16(a, bm, acc[st]);
      }
    }
    const int col = n0 + m;
    const float q = sQ[col], vv = sV[col];
#pragma unroll
    for (int st = 0; st < 4; ++st)
#pragma unroll
      for (int v = 0; v < 8; ++v) ps[st][v] += tanhf(acc[st][v] + q) * vv;
  }
  // reduce over the 16 columns held by each half-wave
#pragma unroll
  for (int st = 0; st < 4; ++st)
#pragma unroll
    for (int v = 0; v < 8; ++v) {
      float x = ps[st][v];
      for (int off = 1; off < 16; off <<= 1) x += __shfl_xor(x, off, 32);
      ps[st][v] = x;
    }
  if (m == 0) {
#pragma unroll
    for (int st = 0; st < 4; ++st)
#pragma unroll
      for (int v = 0; v < 8; ++v) sS[wave][st * 16 + 8 * half + v] = ps[st][v];
  }
  __syncthreads();
  if (tid < 64) {
    float s = Va_b[0];
#pragma unroll
    for (int w = 0; w < 16; ++w) s += sS[w][tid];
    scores[b * CS + sh * 64 + tid] = s;
  }
}

// ---------------------------------------------------------------------------
// K3: softmax over S, context = weights @ enc, x = [embed, context].
// ---------------------------------------------------------------------------
__global__ void softmax_ctx_kernel(const int* __restrict__ input_seq,
                                   const float* __restrict__ scores,
                                   const float* __restrict__ enc,
                                   const float* __restrict__ emb,
                                   float* __restrict__ xv,
                                   float* __restrict__ wout) {
  __shared__ float sSc[CS];
  __shared__ float sRed[CS];
  const int b = blockIdx.x, tid = threadIdx.x;

  if (tid < CS) { sSc[tid] = scores[b * CS + tid]; sRed[tid] = sSc[tid]; }
  __syncthreads();
  for (int s = 64; s > 0; s >>= 1) {
    if (tid < s) sRed[tid] = fmaxf(sRed[tid], sRed[tid + s]);
    __syncthreads();
  }
  const float mx = sRed[0];
  __syncthreads();
  if (tid < CS) { float e = __expf(sSc[tid] - mx); sSc[tid] = e; sRed[tid] = e; }
  __syncthreads();
  for (int s = 64; s > 0; s >>= 1) {
    if (tid < s) sRed[tid] += sRed[tid + s];
    __syncthreads();
  }
  const float denom = sRed[0];
  __syncthreads();
  if (tid < CS) {
    float w = sSc[tid] / denom;
    sSc[tid] = w;
    wout[b * CS + tid] = w;
  }
  __syncthreads();

  const float* eb = enc + (size_t)b * CS * CH;
  for (int h = tid; h < CH; h += 256) {
    float c = 0.0f;
    for (int s = 0; s < CS; ++s) c += sSc[s] * eb[s * CH + h];
    xv[b * (CE + CH) + CE + h] = c;
  }
  const int idx = input_seq[b];
  const float* er = emb + (size_t)idx * CE;
  for (int h = tid; h < CE; h += 256) xv[b * (CE + CH) + h] = er[h];
}

// ---------------------------------------------------------------------------
// K4: gates(64x4096) = x @ W_ih^T + b_ih + h0 @ W_hh^T + b_hh.
// One block per 16-col tile (256 blocks x 128 threads).  [W_ih | W_hh] tile
// staged once as bf16 in LDS (~96 KB) and shared by all 4 M-tile waves, so
// the 48 MB of LSTM weights stream from HBM exactly once.
// ---------------------------------------------------------------------------
__global__ void gates_kernel(const float* __restrict__ xv,
                             const float* __restrict__ h0,
                             const float* __restrict__ W_ih,
                             const float* __restrict__ W_hh,
                             const float* __restrict__ b_ih,
                             const float* __restrict__ b_hh,
                             float* __restrict__ gates) {
  __shared__ __bf16 sB[16][3080];   // [n][0..2047]=W_ih row, [2048..3071]=W_hh row
  const int tid = threadIdx.x;      // 128
  const int n0 = blockIdx.x * 16;

  const float4* s1 = reinterpret_cast<const float4*>(W_ih + (size_t)n0 * (CE + CH));
  for (int i = tid; i < 8192; i += 128) {           // 16 rows x 512 float4
    float4 v = s1[i];
    __bf16* d = &sB[i >> 9][(i & 511) << 2];
    d[0] = (__bf16)v.x; d[1] = (__bf16)v.y; d[2] = (__bf16)v.z; d[3] = (__bf16)v.w;
  }
  const float4* s2 = reinterpret_cast<const float4*>(W_hh + (size_t)n0 * CH);
  for (int i = tid; i < 4096; i += 128) {           // 16 rows x 256 float4
    float4 v = s2[i];
    __bf16* d = &sB[i >> 8][2048 + ((i & 255) << 2)];
    d[0] = (__bf16)v.x; d[1] = (__bf16)v.y; d[2] = (__bf16)v.z; d[3] = (__bf16)v.w;
  }
  __syncthreads();

  const int wave = tid >> 5, lane = tid & 31, m = lane & 15, half = lane >> 4;
  const int row0 = wave * 16;
  v8f acc = {};
  for (int kb = 0; kb < (CE + CH); kb += 32) {
    v16bf a = a_from_f32(xv + (row0 + m) * (CE + CH) + kb + half * 16);
    v16bf b = b_from_lds(&sB[m][kb + half * 16]);
    acc = wmma_bf16(a, b, acc);
  }
  for (int kb = 0; kb < CH; kb += 32) {
    v16bf a = a_from_f32(h0 + (row0 + m) * CH + kb + half * 16);
    v16bf b = b_from_lds(&sB[m][2048 + kb + half * 16]);
    acc = wmma_bf16(a, b, acc);
  }
  const int col = n0 + m;
  const float bias = b_ih[col] + b_hh[col];
#pragma unroll
  for (int v = 0; v < 8; ++v)
    gates[(row0 + v + 8 * half) * (4 * CH) + col] = acc[v] + bias;
}

// ---------------------------------------------------------------------------
// K5: LSTM elementwise.
// ---------------------------------------------------------------------------
__global__ void lstm_kernel(const float* __restrict__ gates,
                            const float* __restrict__ cell,
                            float* __restrict__ h1w,
                            float* __restrict__ h1o,
                            float* __restrict__ c1o) {
  const int i = blockIdx.x * blockDim.x + threadIdx.x;
  if (i >= CB * CH) return;
  const int b = i >> 10, h = i & (CH - 1);
  const float* g = gates + (size_t)b * 4 * CH;
  const float gi = g[h], gf = g[CH + h], gg = g[2 * CH + h], go = g[3 * CH + h];
  const float c0 = cell[i];
  const float c1 = sigmoidf(gf) * c0 + sigmoidf(gi) * tanhf(gg);
  const float h1 = sigmoidf(go) * tanhf(c1);
  h1w[i] = h1; h1o[i] = h1; c1o[i] = c1;
}

// ---------------------------------------------------------------------------
// K6: logits(64x32000) = h1 @ out_w^T + out_b.
// One block per 16-col vocab tile; out_w tile staged to LDS once (async
// global->LDS DMA when available), converted to bf16, reused by 4 waves so
// the 128 MB matrix streams from HBM exactly once.
// ---------------------------------------------------------------------------
__global__ void logits_kernel(const float* __restrict__ h1,
                              const float* __restrict__ out_w,
                              const float* __restrict__ out_b,
                              float* __restrict__ logits) {
#ifdef USE_ASYNC_LDS
  __shared__ float  sF[16384];      // raw f32 tile landed by async DMA (64 KB)
#endif
  __shared__ __bf16 sB[16][1032];
  const int tid = threadIdx.x;      // 128
  const int n0 = blockIdx.x * 16;

  const float4* src = reinterpret_cast<const float4*>(out_w + (size_t)n0 * CH);
#ifdef USE_ASYNC_LDS
  for (int i = tid; i < 4096; i += 128)
    __builtin_amdgcn_global_load_async_to_lds_b128(AS1P(src + i), AS3P(&sF[4 * i]), 0, 0);
  __builtin_amdgcn_s_wait_asynccnt(0);
  __syncthreads();
  for (int i = tid; i < 4096; i += 128) {
    float4 v = reinterpret_cast<const float4*>(sF)[i];
    __bf16* d = &sB[i >> 8][(i & 255) << 2];
    d[0] = (__bf16)v.x; d[1] = (__bf16)v.y; d[2] = (__bf16)v.z; d[3] = (__bf16)v.w;
  }
#else
  for (int i = tid; i < 4096; i += 128) {
    float4 v = src[i];
    __bf16* d = &sB[i >> 8][(i & 255) << 2];
    d[0] = (__bf16)v.x; d[1] = (__bf16)v.y; d[2] = (__bf16)v.z; d[3] = (__bf16)v.w;
  }
#endif
  __syncthreads();

  const int wave = tid >> 5, lane = tid & 31, m = lane & 15, half = lane >> 4;
  const int row0 = wave * 16;
  v8f acc = {};
  for (int kb = 0; kb < CH; kb += 32) {
    v16bf a = a_from_f32(h1 + (row0 + m) * CH + kb + half * 16);
    v16bf b = b_from_lds(&sB[m][kb + half * 16]);
    acc = wmma_bf16(a, b, acc);
  }
  const int col = n0 + m;
  const float bias = out_b[col];
#pragma unroll
  for (int v = 0; v < 8; ++v)
    logits[(size_t)(row0 + v + 8 * half) * CV + col] = acc[v] + bias;
}

// ---------------------------------------------------------------------------
// Host launcher
// ---------------------------------------------------------------------------
extern "C" void kernel_launch(void* const* d_in, const int* in_sizes, int n_in,
                              void* d_out, int out_size, void* d_ws, size_t ws_size,
                              hipStream_t stream) {
  const int*   input_seq = (const int*)  d_in[0];
  const float* hidden    = (const float*)d_in[1];   // (1,B,H) -> h0
  const float* cell      = (const float*)d_in[2];
  const float* enc       = (const float*)d_in[3];
  const float* emb       = (const float*)d_in[4];
  const float* Wa_w      = (const float*)d_in[5];
  const float* Wa_b      = (const float*)d_in[6];
  const float* Ua_w      = (const float*)d_in[7];
  const float* Ua_b      = (const float*)d_in[8];
  const float* Va_w      = (const float*)d_in[9];
  const float* Va_b      = (const float*)d_in[10];
  const float* W_ih      = (const float*)d_in[11];
  const float* W_hh      = (const float*)d_in[12];
  const float* b_ih      = (const float*)d_in[13];
  const float* b_hh      = (const float*)d_in[14];
  const float* out_w     = (const float*)d_in[15];
  const float* out_b     = (const float*)d_in[16];

  float* ws     = (float*)d_ws;
  float* qp     = ws;                 // 64*1024
  float* scores = ws + 65536;         // 64*128
  float* xv     = ws + 73728;         // 64*2048
  float* gates  = ws + 204800;        // 64*4096
  float* h1w    = ws + 466944;        // 64*1024

  float* out    = (float*)d_out;
  float* logits = out;                          // 64*32000
  float* h1o    = out + 2048000;                // 64*1024
  float* c1o    = out + 2113536;                // 64*1024
  float* wout   = out + 2179072;                // 64*128

  qproj_kernel<<<dim3(4, 64), 32, 0, stream>>>(hidden, Wa_w, Wa_b, qp);
  scores_kernel<<<CB * 2, 512, 0, stream>>>(enc, Ua_w, Ua_b, Va_w, Va_b, qp, scores);
  softmax_ctx_kernel<<<CB, 256, 0, stream>>>(input_seq, scores, enc, emb, xv, wout);
  gates_kernel<<<256, 128, 0, stream>>>(xv, hidden, W_ih, W_hh, b_ih, b_hh, gates);
  lstm_kernel<<<(CB * CH + 255) / 256, 256, 0, stream>>>(gates, cell, h1w, h1o, c1o);
  logits_kernel<<<CV / 16, 128, 0, stream>>>(h1w, out_w, out_b, logits);
}
